// Predictor_75917841924330
// MI455X (gfx1250) — matrix operand order repaired and checked by reading, not compiled, not run on hardware
//
#include <hip/hip_runtime.h>
#include <hip/hip_bf16.h>

typedef __attribute__((ext_vector_type(16))) _Float16 v16h;
typedef __attribute__((ext_vector_type(8)))  _Float16 half8;
typedef __attribute__((ext_vector_type(8)))  float    v8f;

// ---------------------------------------------------------------------------
// Elementwise repack kernels
// ---------------------------------------------------------------------------

// NCHW f32 -> NHWC f16, channel-padded to Cp (zeros beyond C)
__global__ void nchw_to_nhwc_f16(const float* __restrict__ in, _Float16* __restrict__ out,
                                 int N, int C, int H, int W, int Cp) {
    size_t i = (size_t)blockIdx.x * blockDim.x + threadIdx.x;
    size_t total = (size_t)N * H * W * Cp;
    if (i >= total) return;
    int c = (int)(i % Cp);
    size_t r = i / Cp;
    int x = (int)(r % W); r /= W;
    int y = (int)(r % H);
    int n = (int)(r / H);
    _Float16 v = (_Float16)0.0f;
    if (c < C) v = (_Float16)in[(((size_t)n * C + c) * H + y) * W + x];
    out[i] = v;
}

// OIHW f32 -> [Cout][T][Cp] f16 (T = KH*KW), channel-padded
__global__ void repack_weight_f16(const float* __restrict__ w, _Float16* __restrict__ out,
                                  int Cout, int Cin, int T, int Cp) {
    size_t i = (size_t)blockIdx.x * blockDim.x + threadIdx.x;
    size_t total = (size_t)Cout * T * Cp;
    if (i >= total) return;
    int c = (int)(i % Cp);
    size_t r = i / Cp;
    int t  = (int)(r % T);
    int co = (int)(r / T);
    _Float16 v = (_Float16)0.0f;
    if (c < Cin) v = (_Float16)w[((size_t)co * Cin + c) * T + t];
    out[i] = v;
}

// ---------------------------------------------------------------------------
// Implicit-GEMM conv using v_wmma_f32_16x16x32_f16.
// GEMM N-dimension = flattened (image, spatial) position p = n*OH*OW + sp,
// tiled as NB consecutive 16-column tiles per wave (zero padding waste when
// P % (16*NB) == 0). The A (weight) fragment is shared by the NB independent
// WMMAs per K-chunk -> 4 back-to-back v_wmma ops, 40B/lane/WMMA.
// X : NHWC f16 [N][H][W][Cp]   (Cp % 32 == 0)
// Wt: [Cout][T][Cp] f16        (Cout % 16 == 0, T = KH*KW)
// Y : NHWC f16 [N][OH][OW][Cout]  == [p][Cout]
// blockDim.x = 256 (8 waves); consecutive waves share the weight tile (coT).
// ---------------------------------------------------------------------------
template <int NB>
__global__ void conv_wmma_nhwc(const _Float16* __restrict__ X,
                               const _Float16* __restrict__ Wt,
                               const float* __restrict__ bias,
                               _Float16* __restrict__ Y,
                               int N, int H, int W, int Cp,
                               int Cout, int OH, int OW,
                               int KH, int KW, int stride, int pad, int relu) {
    const int T = KH * KW;
    const int P = N * OH * OW;
    const int pTiles  = (P + 15) / 16;
    const int pGroups = (pTiles + NB - 1) / NB;
    const int coTiles = Cout / 16;

    int wave = blockIdx.x * (blockDim.x >> 5) + (threadIdx.x >> 5);
    int totalWaves = pGroups * coTiles;
    if (wave >= totalWaves) return;          // wave-uniform; EXEC stays all-ones

    int coT = wave / pGroups;                // adjacent waves share weights
    int pg  = wave % pGroups;

    int lane = threadIdx.x & 31;
    int lm   = lane & 15;
    int lh   = (lane >> 4) & 1;              // K-half select

    // A operand: this lane's weight row (row M of D)
    const _Float16* wrow = Wt + (size_t)(coT * 16 + lm) * T * Cp;
    int coD = coT * 16 + lh * 8;             // D: lane holds co = coD + v

    // Per-subtile position decode (B column of D for this lane)
    int  pj[NB], nj[NB], oyj[NB], oxj[NB];
    bool valid[NB];
#pragma unroll
    for (int j = 0; j < NB; ++j) {
        int p = (pg * NB + j) * 16 + lm;
        valid[j] = p < P;
        int pp = valid[j] ? p : 0;
        int sp = pp % (OH * OW);
        nj[j]  = pp / (OH * OW);
        oyj[j] = sp / OW;
        oxj[j] = sp % OW;
        pj[j]  = p;
    }

    v8f acc[NB];
#pragma unroll
    for (int j = 0; j < NB; ++j)
#pragma unroll
        for (int v = 0; v < 8; ++v) acc[j][v] = bias[coD + v];

    for (int t = 0; t < T; ++t) {
        int kh = t / KW, kw = t % KW;
        const _Float16* wt = wrow + (size_t)t * Cp;

        const _Float16* xr[NB];
        bool inb[NB];
#pragma unroll
        for (int j = 0; j < NB; ++j) {
            int iy = oyj[j] * stride - pad + kh;
            int ix = oxj[j] * stride - pad + kw;
            inb[j] = valid[j] && (iy >= 0) && (iy < H) && (ix >= 0) && (ix < W);
            xr[j] = X + (((size_t)nj[j] * H + (inb[j] ? iy : 0)) * W +
                         (inb[j] ? ix : 0)) * Cp + lh * 16;
        }

        // warm WGP$ for the next tap (global_prefetch_b8)
        if (t + 1 < T) {
            __builtin_prefetch(wrow + (size_t)(t + 1) * Cp, 0, 0);
            if (inb[0]) __builtin_prefetch(xr[0] + Cp, 0, 0);
        }

        for (int c0 = 0; c0 < Cp; c0 += 32) {
            // A: K = {c0+lh*8 .. +7} U {c0+16+lh*8 .. +7}  (ISA 16-bit A layout)
            half8 alo = *(const half8*)(wt + c0 + lh * 8);
            half8 ahi = *(const half8*)(wt + c0 + 16 + lh * 8);
            v16h a = __builtin_shufflevector(alo, ahi,
                0, 1, 2, 3, 4, 5, 6, 7, 8, 9, 10, 11, 12, 13, 14, 15);

            // B_j: K = {c0+lh*8 .. +7} U {c0+16+lh*8 .. +7} at lane's (n,iy,ix)
            v16h bm[NB];
#pragma unroll
            for (int j = 0; j < NB; ++j) {
                v16h b = {};
                if (inb[j]) {
                    half8 b0 = *(const half8*)(xr[j] + c0 - lh * 8);
                    half8 b1 = *(const half8*)(xr[j] + c0 + 16 - lh * 8);
                    b = __builtin_shufflevector(b0, b1,
                        0, 1, 2, 3, 4, 5, 6, 7, 8, 9, 10, 11, 12, 13, 14, 15);
                }
                bm[j] = b;
            }

            // NB independent WMMAs sharing the A fragment (XDL pipelining)
#pragma unroll
            for (int j = 0; j < NB; ++j)
                acc[j] = __builtin_amdgcn_wmma_f32_16x16x32_f16(
                    false, a, false, bm[j], (short)0, acc[j], false, false);
        }
    }

#pragma unroll
    for (int j = 0; j < NB; ++j) {
        if (valid[j]) {
            half8 o;
#pragma unroll
            for (int v = 0; v < 8; ++v) {
                float f = acc[j][v];
                if (relu) f = fmaxf(f, 0.0f);
                o[v] = (_Float16)f;
            }
            *(half8*)(Y + (size_t)pj[j] * Cout + coD) = o;
        }
    }
}

// ---------------------------------------------------------------------------
// Fusion tail kernels (tiny fraction of FLOPs; scalar f32)
// ---------------------------------------------------------------------------

// S[n][sp][544] = concat(bb_feat(4), obj2(256), glob(256), zeros(28)) as f16
__global__ void build_sel_input(const float* __restrict__ bbox,
                                const _Float16* __restrict__ obj2, // [256][49][256]
                                const _Float16* __restrict__ glob, // [49][256]
                                _Float16* __restrict__ S) {
    size_t i = (size_t)blockIdx.x * blockDim.x + threadIdx.x;
    size_t total = (size_t)256 * 49 * 544;
    if (i >= total) return;
    int c  = (int)(i % 544);
    int sp = (int)((i / 544) % 49);
    int n  = (int)(i / (544 * 49));
    _Float16 v = (_Float16)0.0f;
    if (c < 4) {
        float x0 = bbox[n * 4 + 0], y0 = bbox[n * 4 + 1];
        float x1 = bbox[n * 4 + 2], y1 = bbox[n * 4 + 3];
        float f;
        if      (c == 0) f = x0 / 1280.0f;
        else if (c == 1) f = y0 / 720.0f;
        else if (c == 2) f = __logf(1280.0f / (x1 - x0));
        else             f = __logf(720.0f  / (y1 - y0));
        v = (_Float16)f;
    } else if (c < 260) {
        v = obj2[((size_t)n * 49 + sp) * 256 + (c - 4)];
    } else if (c < 516) {
        v = glob[(size_t)sp * 256 + (c - 260)];
    }
    S[i] = v;
}

__global__ void glob_mean_kernel(const _Float16* __restrict__ g, float* __restrict__ gm) {
    int c = threadIdx.x; // 256 threads
    float acc = 0.0f;
    for (int sp = 0; sp < 49; ++sp) acc += (float)g[(size_t)sp * 256 + c];
    gm[c] = acc * (1.0f / 49.0f);
}

// sel3: 1x1 score from [256][3][3][16] f16 with w_sel3 (1,16,3,3)
__global__ void sel3_kernel(const _Float16* __restrict__ ys2,
                            const float* __restrict__ w3, const float* __restrict__ b3,
                            float* __restrict__ sc) {
    int n = threadIdx.x; // 256 threads
    float acc = b3[0];
    for (int t = 0; t < 9; ++t)
        for (int c = 0; c < 16; ++c)
            acc += (float)ys2[((size_t)n * 9 + t) * 16 + c] * w3[c * 9 + t];
    sc[n] = acc;
}

// stable descending top-5 (matches argsort(-s)[:5] tie behavior)
__global__ void top5_kernel(const float* __restrict__ s, int* __restrict__ idx) {
    if (threadIdx.x != 0) return;
    int chosen[5];
    for (int k = 0; k < 5; ++k) {
        float bv = -3.4e38f; int bi = 0;
        for (int i = 0; i < 256; ++i) {
            bool used = false;
            for (int j = 0; j < k; ++j) used |= (chosen[j] == i);
            if (!used && s[i] > bv) { bv = s[i]; bi = i; }
        }
        chosen[k] = bi;
        idx[k] = bi;
    }
}

// final FC head, single block of 256 threads
__global__ void head_kernel(const _Float16* __restrict__ S,   // [256][49][544]
                            const float* __restrict__ gmean,  // [256]
                            const int* __restrict__ idx,      // [5]
                            const float* __restrict__ w1, const float* __restrict__ b1,
                            const float* __restrict__ w2, const float* __restrict__ b2,
                            const float* __restrict__ w3, const float* __restrict__ b3,
                            float* __restrict__ out) {
    __shared__ float xs[5 * 516];
    __shared__ float h1[1280];
    __shared__ float h2[64];
    int tid = threadIdx.x;

    // obj[idx]/5 -> mean over 7x7  (channels 0..259 of S)
    for (int i = tid; i < 5 * 260; i += 256) {
        int k = i / 260, c = i % 260;
        int n = idx[k];
        float acc = 0.0f;
        for (int sp = 0; sp < 49; ++sp)
            acc += (float)S[((size_t)n * 49 + sp) * 544 + c];
        xs[k * 516 + c] = acc * (1.0f / 245.0f);
    }
    // broadcast glob mean (channels 260..515)
    for (int i = tid; i < 5 * 256; i += 256) {
        int k = i / 256, c = i % 256;
        xs[k * 516 + 260 + c] = gmean[c];
    }
    __syncthreads();

    // fc1: [5][516] @ (516,256) -> relu
    for (int i = tid; i < 1280; i += 256) {
        int k = i / 256, j = i % 256;
        float acc = b1[j];
        for (int c = 0; c < 516; ++c) acc += xs[k * 516 + c] * w1[c * 256 + j];
        h1[i] = fmaxf(acc, 0.0f);
    }
    __syncthreads();

    // fc2: [1280] @ (1280,64) -> relu
    if (tid < 64) {
        float acc = b2[tid];
        for (int c = 0; c < 1280; ++c) acc += h1[c] * w2[c * 64 + tid];
        h2[tid] = fmaxf(acc, 0.0f);
    }
    __syncthreads();

    // fc3: [64] @ (64,4)
    if (tid < 4) {
        float acc = b3[tid];
        for (int c = 0; c < 64; ++c) acc += h2[c] * w3[c * 4 + tid];
        out[tid] = acc;
    }
}

// ---------------------------------------------------------------------------
// Launcher
// ---------------------------------------------------------------------------
extern "C" void kernel_launch(void* const* d_in, const int* in_sizes, int n_in,
                              void* d_out, int out_size, void* d_ws, size_t ws_size,
                              hipStream_t stream) {
    (void)in_sizes; (void)n_in; (void)out_size; (void)ws_size;

    const float* glob_feature = (const float*)d_in[0];
    const float* roi_features = (const float*)d_in[1];
    const float* bbox         = (const float*)d_in[2];
    const float* w_glob1 = (const float*)d_in[3];  const float* b_glob1 = (const float*)d_in[4];
    const float* w_glob2 = (const float*)d_in[5];  const float* b_glob2 = (const float*)d_in[6];
    const float* w_obj1  = (const float*)d_in[7];  const float* b_obj1  = (const float*)d_in[8];
    const float* w_obj2  = (const float*)d_in[9];  const float* b_obj2  = (const float*)d_in[10];
    const float* w_sel1  = (const float*)d_in[11]; const float* b_sel1  = (const float*)d_in[12];
    const float* w_sel2  = (const float*)d_in[13]; const float* b_sel2  = (const float*)d_in[14];
    const float* w_sel3  = (const float*)d_in[15]; const float* b_sel3  = (const float*)d_in[16];
    const float* w_fc1   = (const float*)d_in[17]; const float* b_fc1   = (const float*)d_in[18];
    const float* w_fc2   = (const float*)d_in[19]; const float* b_fc2   = (const float*)d_in[20];
    const float* w_fc3   = (const float*)d_in[21]; const float* b_fc3   = (const float*)d_in[22];
    float* out = (float*)d_out;

    // workspace carve-up (256B aligned)
    unsigned char* ws = (unsigned char*)d_ws;
    size_t off = 0;
    auto alloc = [&](size_t bytes) -> void* {
        void* p = ws + off;
        off = (off + bytes + 255) & ~(size_t)255;
        return p;
    };
    _Float16* xg   = (_Float16*)alloc((size_t)28 * 28 * 1024 * 2);        // glob NHWC
    _Float16* wg1  = (_Float16*)alloc((size_t)512 * 16 * 1024 * 2);
    _Float16* yg1  = (_Float16*)alloc((size_t)14 * 14 * 512 * 2);
    _Float16* wg2  = (_Float16*)alloc((size_t)256 * 16 * 512 * 2);
    _Float16* yg2  = (_Float16*)alloc((size_t)49 * 256 * 2);              // glob 7x7x256
    _Float16* xr   = (_Float16*)alloc((size_t)256 * 49 * 2048 * 2);       // roi NHWC
    _Float16* wo1  = (_Float16*)alloc((size_t)1024 * 9 * 2048 * 2);
    _Float16* yo1  = (_Float16*)alloc((size_t)256 * 49 * 1024 * 2);
    _Float16* wo2  = (_Float16*)alloc((size_t)256 * 9 * 1024 * 2);
    _Float16* yo2  = (_Float16*)alloc((size_t)256 * 49 * 256 * 2);
    _Float16* Sbuf = (_Float16*)alloc((size_t)256 * 49 * 544 * 2);
    _Float16* ws1  = (_Float16*)alloc((size_t)256 * 9 * 544 * 2);
    _Float16* ys1  = (_Float16*)alloc((size_t)256 * 25 * 256 * 2);
    _Float16* ws2  = (_Float16*)alloc((size_t)16 * 9 * 256 * 2);
    _Float16* ys2  = (_Float16*)alloc((size_t)256 * 9 * 16 * 2);
    float*    sc   = (float*)alloc(256 * sizeof(float));
    int*      idx5 = (int*)alloc(5 * sizeof(int));
    float*    gm   = (float*)alloc(256 * sizeof(float));

    auto ew = [&](size_t total) { return (unsigned)((total + 255) / 256); };

    auto launch_conv1 = [&](const _Float16* X, const _Float16* Wt, const float* bias, _Float16* Y,
                            int N, int H, int W, int Cp, int Cout, int OH, int OW,
                            int KH, int KW, int strd, int pad, int relu) {
        int pTiles = (N * OH * OW + 15) / 16;
        int waves  = pTiles * (Cout / 16);
        hipLaunchKernelGGL((conv_wmma_nhwc<1>), dim3((unsigned)((waves + 7) / 8)), dim3(256), 0, stream,
                           X, Wt, bias, Y, N, H, W, Cp, Cout, OH, OW, KH, KW, strd, pad, relu);
    };
    auto launch_conv4 = [&](const _Float16* X, const _Float16* Wt, const float* bias, _Float16* Y,
                            int N, int H, int W, int Cp, int Cout, int OH, int OW,
                            int KH, int KW, int strd, int pad, int relu) {
        int pTiles  = (N * OH * OW + 15) / 16;
        int pGroups = (pTiles + 3) / 4;
        int waves   = pGroups * (Cout / 16);
        hipLaunchKernelGGL((conv_wmma_nhwc<4>), dim3((unsigned)((waves + 7) / 8)), dim3(256), 0, stream,
                           X, Wt, bias, Y, N, H, W, Cp, Cout, OH, OW, KH, KW, strd, pad, relu);
    };

    // --- global-feature branch ---
    hipLaunchKernelGGL(nchw_to_nhwc_f16, dim3(ew((size_t)28 * 28 * 1024)), dim3(256), 0, stream,
                       glob_feature, xg, 1, 1024, 28, 28, 1024);
    hipLaunchKernelGGL(repack_weight_f16, dim3(ew((size_t)512 * 16 * 1024)), dim3(256), 0, stream,
                       w_glob1, wg1, 512, 1024, 16, 1024);
    launch_conv1(xg, wg1, b_glob1, yg1, 1, 28, 28, 1024, 512, 14, 14, 4, 4, 2, 1, 1);
    hipLaunchKernelGGL(repack_weight_f16, dim3(ew((size_t)256 * 16 * 512)), dim3(256), 0, stream,
                       w_glob2, wg2, 256, 512, 16, 512);
    launch_conv1(yg1, wg2, b_glob2, yg2, 1, 14, 14, 512, 256, 7, 7, 4, 4, 2, 1, 1);
    hipLaunchKernelGGL(glob_mean_kernel, dim3(1), dim3(256), 0, stream, yg2, gm);

    // --- object branch (dominant FLOPs) ---
    hipLaunchKernelGGL(nchw_to_nhwc_f16, dim3(ew((size_t)256 * 49 * 2048)), dim3(256), 0, stream,
                       roi_features, xr, 256, 2048, 7, 7, 2048);
    hipLaunchKernelGGL(repack_weight_f16, dim3(ew((size_t)1024 * 9 * 2048)), dim3(256), 0, stream,
                       w_obj1, wo1, 1024, 2048, 9, 2048);
    launch_conv4(xr, wo1, b_obj1, yo1, 256, 7, 7, 2048, 1024, 7, 7, 3, 3, 1, 1, 1);
    hipLaunchKernelGGL(repack_weight_f16, dim3(ew((size_t)256 * 9 * 1024)), dim3(256), 0, stream,
                       w_obj2, wo2, 256, 1024, 9, 1024);
    launch_conv4(yo1, wo2, b_obj2, yo2, 256, 7, 7, 1024, 256, 7, 7, 3, 3, 1, 1, 1);

    // --- selection branch ---
    hipLaunchKernelGGL(build_sel_input, dim3(ew((size_t)256 * 49 * 544)), dim3(256), 0, stream,
                       bbox, yo2, yg2, Sbuf);
    hipLaunchKernelGGL(repack_weight_f16, dim3(ew((size_t)256 * 9 * 544)), dim3(256), 0, stream,
                       w_sel1, ws1, 256, 516, 9, 544);
    launch_conv4(Sbuf, ws1, b_sel1, ys1, 256, 7, 7, 544, 256, 5, 5, 3, 3, 1, 0, 1);
    hipLaunchKernelGGL(repack_weight_f16, dim3(ew((size_t)16 * 9 * 256)), dim3(256), 0, stream,
                       w_sel2, ws2, 16, 256, 9, 256);
    launch_conv4(ys1, ws2, b_sel2, ys2, 256, 5, 5, 256, 16, 3, 3, 3, 3, 1, 0, 1);
    hipLaunchKernelGGL(sel3_kernel, dim3(1), dim3(256), 0, stream, ys2, w_sel3, b_sel3, sc);
    hipLaunchKernelGGL(top5_kernel, dim3(1), dim3(32), 0, stream, sc, idx5);

    // --- FC head ---
    hipLaunchKernelGGL(head_kernel, dim3(1), dim3(256), 0, stream,
                       Sbuf, gm, idx5, w_fc1, b_fc1, w_fc2, b_fc2, w_fc3, b_fc3, out);
}
